// WTConv2d_CUDA_46514495816358
// MI455X (gfx1250) — matrix operand order repaired and checked by reading, not compiled, not run on hardware
//
#include <hip/hip_runtime.h>
#include <hip/hip_bf16.h>

// WTConv2d for MI455X (gfx1250): Haar DWT -> implicit-GEMM 3x3 conv via
// v_wmma_f32_16x16x32_f16 -> Haar IDWT chain. The three convs are GEMMs
// (M=36864/9216/2304 pixels, N=256, K=2304) totalling ~57 GFLOP against
// ~100 MB of traffic (L2-resident), i.e. firmly compute-bound on the
// matrix pipes -> f16 WMMA with f32 accumulation is the right path.

typedef _Float16 v8h  __attribute__((ext_vector_type(8)));
typedef _Float16 v16h __attribute__((ext_vector_type(16)));
typedef float    v8f  __attribute__((ext_vector_type(8)));

#define CHN  256          // conv channels (4*64)
#define KTOT 2304         // 9 taps * 256 channels

// ---------------------------------------------------------------------------
// Zero a small f16 region (zero page for padded conv taps).
// ---------------------------------------------------------------------------
__global__ __launch_bounds__(256)
void wt_zero16(_Float16* __restrict__ p, int n) {
    int i = blockIdx.x * 256 + threadIdx.x;
    if (i < n) p[i] = (_Float16)0.0f;
}

// ---------------------------------------------------------------------------
// Weight repack: w[co][ci][kh][kw] f32  ->  wp[co][tap*256+ci] f16
// K-contiguous per output channel so B fragments are contiguous 32B runs.
// ---------------------------------------------------------------------------
__global__ __launch_bounds__(256)
void wt_repack_w(const float* __restrict__ w, _Float16* __restrict__ wp) {
    int idx = blockIdx.x * 256 + threadIdx.x;
    if (idx >= CHN * CHN * 9) return;
    int co  = idx / (CHN * 9);
    int r   = idx - co * (CHN * 9);
    int ci  = r / 9;
    int tap = r - ci * 9;                       // kh*3+kw
    wp[(size_t)co * KTOT + tap * CHN + ci] = (_Float16)w[idx];
}

// ---------------------------------------------------------------------------
// Haar DWT: in (f32, arbitrary strides) -> out NHWC f16 [B][H2][W2][256]
// channel = quad*64 + c, quad in {ll,lh,hl,hh}; scale s^2 = 0.25
// ---------------------------------------------------------------------------
__global__ __launch_bounds__(256)
void wt_dwt(const float* __restrict__ in,
            long sb, long sc, long sh, long sw,
            _Float16* __restrict__ out,
            int H2, int W2, int total) {
    int idx = blockIdx.x * 256 + threadIdx.x;
    if (idx >= total) return;
    int c  = idx & 63;
    int t  = idx >> 6;
    int w2 = t % W2; t /= W2;
    int h2 = t % H2;
    int b  = t / H2;
    const float* p = in + (size_t)b * sb + (size_t)c * sc
                        + (size_t)(2 * h2) * sh + (size_t)(2 * w2) * sw;
    float va = p[0], vb = p[sw], vc = p[sh], vd = p[sh + sw];
    const float s2 = 0.25f;   // DWT_SCALE^2
    float ll = s2 * (va + vb + vc + vd);
    float lh = s2 * (va + vb - vc - vd);
    float hl = s2 * (va - vb + vc - vd);
    float hh = s2 * (va - vb - vc + vd);
    _Float16* o = out + ((size_t)((b * H2 + h2) * W2 + w2)) * CHN + c;
    o[0]   = (_Float16)ll;
    o[64]  = (_Float16)lh;
    o[128] = (_Float16)hl;
    o[192] = (_Float16)hh;
}

// ---------------------------------------------------------------------------
// Implicit-GEMM 3x3 conv, pad 1, stride 1, 256->256 channels.
//   act : NHWC f16 [B][H][W][256]      (A matrix, M = pixels, K = tap*256+ci)
//   wp  : f16 [256][2304]              (B matrix, N = out channel)
//   zpg : 256-halfword zero page       (branchless padding)
//   out : NHWC f32 [B][H][W][256]
//
// Workgroup = 8 waves = 128 pixels x 256 channels.
// Wave = 64 px x 64 ch = 4x4 tiles of v_wmma_f32_16x16x32_f16
// (16 WMMA per K-chunk vs 8 A-loads + 4 B-loads).
//
// Loop structure: tap-outer (9) x K-chunk inner (8, fully unrolled) so all
// inner loads are base + immediate-offset; out-of-image taps select the
// zero page instead of branching (EXEC stays all-ones for WMMA).
//
// Fragment layouts (wave32; A/B use contiguous k, same order in both):
//   A 16x32 f16 : kb = lane<16 ? 0 : 16 ; halves [kb,kb+16) contiguous
//   B 32x16 f16 : koff = lane<16 ? 0 : 16 ; halves [koff,koff+16) contiguous
//   C/D 16x16   : lane column N = lane&15 ; VGPR r -> M = r (+8 lanes>=16)
// ---------------------------------------------------------------------------
__global__ __launch_bounds__(256)
void wt_conv_wmma(const _Float16* __restrict__ act,
                  const _Float16* __restrict__ wp,
                  const _Float16* __restrict__ zpg,
                  float* __restrict__ out,
                  int H, int W) {
    const int lane = threadIdx.x & 31;
    const int wave = threadIdx.x >> 5;
    const int wpix = wave & 1;          // pixel half of the 128-px block
    const int wn   = wave >> 1;         // 64-channel quarter (0..3)
    const int l15  = lane & 15;
    const int kb   = (lane < 16) ? 0 : 16;   // A-fragment K base
    const int koff = (lane < 16) ? 0 : 16;   // B-fragment K base
    const int HW   = H * W;
    const int pixBase = blockIdx.x * 128 + wpix * 64;
    const int nBase   = wn * 64;

    // per-lane pixel coordinates for the 4 pixel tiles (M rows)
    int pb[4], poh[4], pow_[4];
#pragma unroll
    for (int t = 0; t < 4; ++t) {
        int m  = pixBase + t * 16 + l15;
        int b  = m / HW;
        int r  = m - b * HW;
        int oh = r / W;
        pb[t]   = b;
        poh[t]  = oh;
        pow_[t] = r - oh * W;
    }

    // per-lane weight row pointers (fixed across the whole K loop)
    const _Float16* brow[4];
#pragma unroll
    for (int u = 0; u < 4; ++u)
        brow[u] = wp + (size_t)(nBase + u * 16 + l15) * KTOT + koff;

    v8f zero8 = {};
    v8f acc[4][4];
#pragma unroll
    for (int t = 0; t < 4; ++t)
#pragma unroll
        for (int u = 0; u < 4; ++u) acc[t][u] = zero8;

#pragma unroll 1
    for (int tap = 0; tap < 9; ++tap) {
        const int kh = tap / 3;
        const int kw = tap - kh * 3;

        // branchless im2col bases: invalid taps read the zero page
        const _Float16* abase[4];
#pragma unroll
        for (int t = 0; t < 4; ++t) {
            int ih = poh[t] + kh - 1;
            int iw = pow_[t] + kw - 1;
            bool ok = ((unsigned)ih < (unsigned)H) & ((unsigned)iw < (unsigned)W);
            abase[t] = ok ? act + ((size_t)((pb[t] * H + ih) * W + iw)) * CHN + kb
                          : zpg + kb;
        }
        const int tapOff = tap * CHN;
        // prefetch next tap's weight row (global_prefetch_b8)
        if (tap < 8)
            __builtin_prefetch((const void*)(brow[0] + tapOff + CHN), 0, 1);

#pragma unroll
        for (int ck = 0; ck < 8; ++ck) {       // 8 K-chunks of 32 per tap
            const int ci0 = ck * 32;
            v16h bfrag[4];
#pragma unroll
            for (int u = 0; u < 4; ++u)
                bfrag[u] = *(const v16h*)(brow[u] + tapOff + ci0);
#pragma unroll
            for (int t = 0; t < 4; ++t) {
                v8h lo = *(const v8h*)(abase[t] + ci0);
                v8h hi = *(const v8h*)(abase[t] + ci0 + 8);
                v16h afrag = __builtin_shufflevector(lo, hi,
                        0, 1, 2, 3, 4, 5, 6, 7, 8, 9, 10, 11, 12, 13, 14, 15);
#pragma unroll
                for (int u = 0; u < 4; ++u) {
                    acc[t][u] = __builtin_amdgcn_wmma_f32_16x16x32_f16(
                        false, afrag, false, bfrag[u],
                        (short)0, acc[t][u], false, false);
                }
            }
        }
    }

    // D layout: lane column N = lane&15; VGPR r -> M = r (+8 for lanes 16..31)
    const int mOff = (lane < 16) ? 0 : 8;
#pragma unroll
    for (int t = 0; t < 4; ++t) {
#pragma unroll
        for (int u = 0; u < 4; ++u) {
            int n = nBase + u * 16 + l15;
#pragma unroll
            for (int r = 0; r < 8; ++r) {
                int m = pixBase + t * 16 + mOff + r;
                out[(size_t)m * CHN + n] = acc[t][u][r];
            }
        }
    }
}

// ---------------------------------------------------------------------------
// Haar IDWT (t^2 = 1):
//   ll    : f32 with strides (NHWC slice of conv out, or NCHW intermediate)
//   hi    : NHWC f32 conv output; lh/hl/hh at channels 64/128/192 + c
//   out   : NCHW f32 [B][64][2H2][2W2] ; optional bias add (final level)
// ---------------------------------------------------------------------------
__global__ __launch_bounds__(256)
void wt_idwt(const float* __restrict__ ll,
             long lsb, long lsc, long lsh, long lsw,
             const float* __restrict__ hi,
             float* __restrict__ out,
             const float* __restrict__ bias,
             int H2, int W2, int total) {
    int idx = blockIdx.x * 256 + threadIdx.x;
    if (idx >= total) return;
    int w2 = idx % W2; int t = idx / W2;
    int h2 = t % H2;   t /= H2;
    int c  = t & 63;
    int b  = t >> 6;

    float vll = ll[(size_t)b * lsb + (size_t)c * lsc
                   + (size_t)h2 * lsh + (size_t)w2 * lsw];
    const float* hp = hi + ((size_t)((b * H2 + h2) * W2 + w2)) * CHN;
    float vlh = hp[64 + c], vhl = hp[128 + c], vhh = hp[192 + c];

    float oa = vll + vlh + vhl + vhh;
    float ob = vll + vlh - vhl - vhh;
    float oc = vll - vlh + vhl - vhh;
    float od = vll - vlh - vhl + vhh;
    float bi = bias ? bias[c] : 0.0f;

    int Wf = 2 * W2, Hf = 2 * H2;
    float* o = out + (((size_t)(b * 64 + c) * Hf + 2 * h2) * Wf + 2 * w2);
    o[0]      = oa + bi;
    o[1]      = ob + bi;
    o[Wf]     = oc + bi;
    o[Wf + 1] = od + bi;
}

// ---------------------------------------------------------------------------
extern "C" void kernel_launch(void* const* d_in, const int* in_sizes, int n_in,
                              void* d_out, int out_size, void* d_ws, size_t ws_size,
                              hipStream_t stream) {
    const float* x    = (const float*)d_in[0];   // (4,64,192,192)
    const float* w0   = (const float*)d_in[1];   // (256,256,3,3)
    const float* w1   = (const float*)d_in[2];
    const float* w2   = (const float*)d_in[3];
    const float* bias = (const float*)d_in[4];   // (64,)

    const int B = 4;
    const int H0 = 96, H1 = 48, H2v = 24;        // post-DWT spatial per level
    const size_t np0 = (size_t)B * H0 * H0;      // 36864 pixels
    const size_t np1 = (size_t)B * H1 * H1;      // 9216
    const size_t np2 = (size_t)B * H2v * H2v;    // 2304

    size_t off = 0;
    char* ws = (char*)d_ws;
    auto alloc = [&](size_t bytes) -> void* {
        void* p = ws + off;
        off += (bytes + 255) & ~(size_t)255;
        return p;
    };

    _Float16* d0  = (_Float16*)alloc(np0 * CHN * 2);   // 18.9 MB (DWT acts)
    _Float16* d1  = (_Float16*)alloc(np1 * CHN * 2);
    _Float16* d2  = (_Float16*)alloc(np2 * CHN * 2);
    float*    y0  = (float*)  alloc(np0 * CHN * 4);    // 37.7 MB (conv outs)
    float*    y1  = (float*)  alloc(np1 * CHN * 4);
    float*    y2  = (float*)  alloc(np2 * CHN * 4);
    _Float16* wp0 = (_Float16*)alloc((size_t)CHN * KTOT * 2);
    _Float16* wp1 = (_Float16*)alloc((size_t)CHN * KTOT * 2);
    _Float16* wp2 = (_Float16*)alloc((size_t)CHN * KTOT * 2);
    _Float16* zpg = (_Float16*)alloc(256 * 2);         // zero page for padding
    // IDWT temporaries alias the (now dead) d0 region
    float* t2 = (float*)d0;                            // (4,64,48,48)
    float* t1 = (float*)((char*)d0 + (4u << 20));      // (4,64,96,96)

    // --- zero page + weight repack (f32 -> WMMA-layout f16) ---
    wt_zero16<<<1, 256, 0, stream>>>(zpg, 256);
    {
        int n = CHN * CHN * 9;
        wt_repack_w<<<(n + 255) / 256, 256, 0, stream>>>(w0, wp0);
        wt_repack_w<<<(n + 255) / 256, 256, 0, stream>>>(w1, wp1);
        wt_repack_w<<<(n + 255) / 256, 256, 0, stream>>>(w2, wp2);
    }

    // --- level 0: DWT(x NCHW) -> conv ---
    {
        int tot = (int)(np0 * 64);
        wt_dwt<<<(tot + 255) / 256, 256, 0, stream>>>(
            x, (long)64 * 192 * 192, (long)192 * 192, 192, 1, d0, H0, H0, tot);
        wt_conv_wmma<<<(unsigned)(np0 / 128), 256, 0, stream>>>(
            d0, wp0, zpg, y0, H0, H0);
    }
    // --- level 1: DWT(y0 NHWC ch0..63) -> conv ---
    {
        int tot = (int)(np1 * 64);
        wt_dwt<<<(tot + 255) / 256, 256, 0, stream>>>(
            y0, (long)H0 * H0 * CHN, 1, (long)H0 * CHN, CHN, d1, H1, H1, tot);
        wt_conv_wmma<<<(unsigned)(np1 / 128), 256, 0, stream>>>(
            d1, wp1, zpg, y1, H1, H1);
    }
    // --- level 2 ---
    {
        int tot = (int)(np2 * 64);
        wt_dwt<<<(tot + 255) / 256, 256, 0, stream>>>(
            y1, (long)H1 * H1 * CHN, 1, (long)H1 * CHN, CHN, d2, H2v, H2v, tot);
        wt_conv_wmma<<<(unsigned)(np2 / 128), 256, 0, stream>>>(
            d2, wp2, zpg, y2, H2v, H2v);
    }

    // --- inverse chain ---
    {   // idwt level2: ll = y2 NHWC ch0..63 -> t2 NCHW (4,64,48,48)
        int tot = (int)(np2 * 64);
        wt_idwt<<<(tot + 255) / 256, 256, 0, stream>>>(
            y2, (long)H2v * H2v * CHN, 1, (long)H2v * CHN, CHN,
            y2, t2, nullptr, H2v, H2v, tot);
    }
    {   // idwt level1: ll = t2 NCHW -> t1 NCHW (4,64,96,96)
        int tot = (int)(np1 * 64);
        wt_idwt<<<(tot + 255) / 256, 256, 0, stream>>>(
            t2, (long)64 * H1 * H1, (long)H1 * H1, H1, 1,
            y1, t1, nullptr, H1, H1, tot);
    }
    {   // idwt level0: ll = t1 NCHW -> d_out NCHW (4,64,192,192) + bias
        int tot = (int)(np0 * 64);
        wt_idwt<<<(tot + 255) / 256, 256, 0, stream>>>(
            t1, (long)64 * H0 * H0, (long)H0 * H0, H0, 1,
            y0, (float*)d_out, bias, H0, H0, tot);
    }
}
